// SAM_7103875907976
// MI455X (gfx1250) — matrix operand order repaired and checked by reading, not compiled, not run on hardware
//
#include <hip/hip_runtime.h>

#define C_CH  256
#define HEADS 8
#define DH    32
#define MM    4096   // H_SP * W_SP

#if __has_builtin(__builtin_amdgcn_global_load_async_to_lds_b128) && \
    __has_builtin(__builtin_amdgcn_s_wait_asynccnt)
#define USE_ASYNC_LDS 1
#else
#define USE_ASYNC_LDS 0
#endif

typedef __attribute__((ext_vector_type(16))) _Float16 v16h;
typedef __attribute__((ext_vector_type(8)))  _Float16 v8h;
typedef __attribute__((ext_vector_type(4)))  _Float16 v4h;
typedef __attribute__((ext_vector_type(8)))  float    v8f;

typedef int i32x4 __attribute__((vector_size(16)));
typedef __attribute__((address_space(1))) i32x4* gptr_b128;   // global (AS1)
typedef __attribute__((address_space(3))) i32x4* lptr_b128;   // LDS (AS3)

union Frag16 { v16h h; uint4 u[2]; };
union Half8  { v8h  h; uint4 u;    };
union Half4  { v4h  h; uint2 u;    };
union F8     { v8f  f; float4 q[2]; };

static __device__ __forceinline__ v8f wmma_f16(v16h a, v16h b, v8f c) {
  // 8 args: (neg_a, A, neg_b, B, c_mod, C, reuse_a, reuse_b)
  return __builtin_amdgcn_wmma_f32_16x16x32_f16(false, a, false, b, (short)0, c,
                                                false, false);
}

#if USE_ASYNC_LDS
static __device__ __forceinline__ void async_cp16(const void* g, void* l) {
  __builtin_amdgcn_global_load_async_to_lds_b128(
      (gptr_b128)(size_t)g, (lptr_b128)(unsigned)(size_t)l, 0, 0);
}
#endif

// ---------------------------------------------------------------------------
// Kernel 1: fixed 2D sinusoidal positional encoding table, f16, layout (H,M,DH)
// ---------------------------------------------------------------------------
__global__ __launch_bounds__(256) void pe_kernel(_Float16* __restrict__ Rt) {
  int idx = blockIdx.x * 256 + threadIdx.x;      // (h*MM + m)*DH + d
  int d = idx & (DH - 1);
  int m = (idx >> 5) & (MM - 1);
  int h = idx >> 17;
  int c = h * DH + d;
  const int half = C_CH / 2;                     // 128
  int cc, pos;
  if (c < half) { cc = c;         pos = m >> 6; }   // height channels
  else          { cc = c - half;  pos = m & 63; }   // width channels
  int t = cc >> 1;
  float dv  = __expf((float)(2 * t) * (-9.210340371976184f / (float)half));
  float ang = (float)pos * dv;
  float val = (cc & 1) ? __cosf(ang) : __sinf(ang);
  Rt[idx] = (_Float16)val;
}

// ---------------------------------------------------------------------------
// Kernel 2: projections.  One block per 16-wide m-tile, 8 waves (wave = head).
//   Vt (B,H,M,DH) f16, scaled by log2(e)/sqrt(DH)  [A-frag layout for scores]
//   Kt (B,H,M,DH) f16, = Z + R                     [B-frag layout for scores]
//   Qd (B,H,DH,M) f16, n permuted inside 64-blocks [B-frag layout for PV]
//     position p within a 64-block holds n = (p&3)*16 + (p>>2)
// ---------------------------------------------------------------------------
static __device__ __forceinline__ v16h load_w_a(const float* __restrict__ W,
                                                int o0, int cb, int lane, bool lo) {
  // A-matrix 16x32 f16: lane<16 -> K {0..7,16..23}; lane>=16 -> K {8..15,24..31}
  const float* r = W + (size_t)(o0 + (lane & 15)) * C_CH + cb + (lo ? 0 : 8);
  Frag16 f;
#pragma unroll
  for (int i = 0; i < 8; ++i) f.h[i]     = (_Float16)r[i];
#pragma unroll
  for (int i = 0; i < 8; ++i) f.h[8 + i] = (_Float16)r[16 + i];
  return f.h;
}
static __device__ __forceinline__ v16h load_w_b(const float* __restrict__ W,
                                                int o0, int cb, int lane, bool lo) {
  // B-matrix 32x16 f16: lane<16 -> K 0..15; lane>=16 -> K 16..31 (N = lane&15)
  const float* r = W + (size_t)(o0 + (lane & 15)) * C_CH + cb + (lo ? 0 : 16);
  Frag16 f;
#pragma unroll
  for (int i = 0; i < 16; ++i) f.h[i] = (_Float16)r[i];
  return f.h;
}

__global__ __launch_bounds__(256) void proj_kernel(
    const float* __restrict__ x,  const float* __restrict__ Wv,
    const float* __restrict__ Wz, const float* __restrict__ Wq,
    const _Float16* __restrict__ Rt,
    _Float16* __restrict__ Vt, _Float16* __restrict__ Kt,
    _Float16* __restrict__ Qd) {
  __shared__ _Float16 xT[16][32];      // x^T tile: (m, c) f16

  const int tid   = threadIdx.x;
  const int lane  = tid & 31;
  const int w     = tid >> 5;          // wave id == head id
  const bool lo   = lane < 16;
  const int mbase = blockIdx.x * 16;
  const int b     = blockIdx.y;

  v8f aV0 = {0,0,0,0,0,0,0,0}, aV1 = {0,0,0,0,0,0,0,0};
  v8f aK0 = {0,0,0,0,0,0,0,0}, aK1 = {0,0,0,0,0,0,0,0};
  v8f aQ0 = {0,0,0,0,0,0,0,0}, aQ1 = {0,0,0,0,0,0,0,0};

  const int o0 = w * 32, o1 = w * 32 + 16;

  for (int cb = 0; cb < C_CH; cb += 32) {
    __syncthreads();                       // protect xT WAR
#pragma unroll
    for (int rep = 0; rep < 2; ++rep) {    // 256 thr x 2 = 512 elements
      int cl = (tid >> 4) + 16 * rep;
      int ml = tid & 15;
      xT[ml][cl] = (_Float16)x[((size_t)b * C_CH + cb + cl) * MM + mbase + ml];
    }
    __syncthreads();

    Frag16 bx, ax;
    {
      const _Float16* row = &xT[lane & 15][0];
      int boff = lo ? 0 : 8;
      bx.u[0] = *(const uint4*)(row + boff);
      bx.u[1] = *(const uint4*)(row + boff + 16);
      int aoff = lo ? 0 : 16;
      ax.u[0] = *(const uint4*)(row + aoff);
      ax.u[1] = *(const uint4*)(row + aoff + 8);
    }

    aV0 = wmma_f16(load_w_a(Wv, o0, cb, lane, lo), bx.h, aV0);
    aV1 = wmma_f16(load_w_a(Wv, o1, cb, lane, lo), bx.h, aV1);
    aK0 = wmma_f16(load_w_a(Wz, o0, cb, lane, lo), bx.h, aK0);
    aK1 = wmma_f16(load_w_a(Wz, o1, cb, lane, lo), bx.h, aK1);
    aQ0 = wmma_f16(ax.h, load_w_b(Wq, o0, cb, lane, lo), aQ0);
    aQ1 = wmma_f16(ax.h, load_w_b(Wq, o1, cb, lane, lo), aQ1);
  }

  const int h = w;
  const float scale = 0.25504377371566784f;   // log2(e)/sqrt(DH)

  // V and K tiles: C/D layout lane=N=m, VGPR r -> o row
  {
    int m = mbase + (lane & 15);
    size_t base = ((size_t)(b * HEADS + h) * MM + m) * DH;
#pragma unroll
    for (int j = 0; j < 2; ++j) {
      int d0 = j * 16 + (lo ? 0 : 8);
      const v8f& av = j ? aV1 : aV0;
      const v8f& ak = j ? aK1 : aK0;
      Half8 pv, pk, rr;
      rr.u = *(const uint4*)&Rt[((size_t)h * MM + m) * DH + d0];
#pragma unroll
      for (int r = 0; r < 8; ++r) {
        pv.h[r] = (_Float16)(av[r] * scale);
        pk.h[r] = (_Float16)(ak[r] + (float)rr.h[r]);
      }
      *(uint4*)&Vt[base + d0] = pv.u;
      *(uint4*)&Kt[base + d0] = pk.u;
    }
  }
  // Q tile: lane = d, VGPR r -> m; store with per-64-block permutation
  {
    int m0 = mbase + (lo ? 0 : 8);
#pragma unroll
    for (int j = 0; j < 2; ++j) {
      int d = j * 16 + (lane & 15);
      const v8f& aq = j ? aQ1 : aQ0;
      size_t rowb = ((size_t)(b * HEADS + h) * DH + d) * MM;
#pragma unroll
      for (int r = 0; r < 8; ++r) {
        int m = m0 + r;
        int p = ((m & 15) << 2) | ((m >> 4) & 3);   // col*4 + t
        Qd[rowb + (size_t)(m & ~63) + p] = (_Float16)aq[r];
      }
    }
  }
}

// ---------------------------------------------------------------------------
// Kernel 3: flash attention.  Block = 4 waves (one 64-query block, same b,h).
// K/Q 64-n chunks double-buffered in LDS via async global->LDS copies.
// scores = V^T K (K-dim = DH = 32 = one WMMA), online softmax (base 2) over n,
// P transposed through per-wave LDS with packed b64 stores (permuted n order).
// ---------------------------------------------------------------------------
__global__ __launch_bounds__(128) void attn_kernel(
    const _Float16* __restrict__ Vt, const _Float16* __restrict__ Kt,
    const _Float16* __restrict__ Qd, float* __restrict__ out) {
  __shared__ _Float16 Kl[2][64 * 32];   // (n, d) chunk, 4 KB each
  __shared__ _Float16 Ql[2][32 * 64];   // (d, n) chunk (permuted n), 4 KB each
  __shared__ _Float16 P2[4][16 * 64];   // per-wave P tile (m, p)

  const int tid   = threadIdx.x;
  const int lane  = tid & 31;
  const int w     = tid >> 5;
  const bool lo   = lane < 16;
  const int mtile = blockIdx.x * 64 + w * 16;
  const int h     = blockIdx.y;
  const int b     = blockIdx.z;

  const _Float16* vbase = Vt + (size_t)(b * HEADS + h) * MM * DH;
  const _Float16* kbase = Kt + (size_t)(b * HEADS + h) * MM * DH;
  const _Float16* qbase = Qd + (size_t)(b * HEADS + h) * DH * MM;

  Frag16 av;   // V A-frag: rows m, K = d
  {
    const _Float16* r = vbase + (size_t)(mtile + (lane & 15)) * DH + (lo ? 0 : 16);
    av.u[0] = *(const uint4*)(r);
    av.u[1] = *(const uint4*)(r + 8);
  }

  v8f o0 = {0,0,0,0,0,0,0,0}, o1 = {0,0,0,0,0,0,0,0};
  float rmax[8], rsum[8];
#pragma unroll
  for (int r = 0; r < 8; ++r) { rmax[r] = -1e30f; rsum[r] = 0.0f; }

  // ---- chunk staging: K rows are a 4 KB linear copy; Q is 32 rows x 128 B
  auto stage = [&](int buf, int nb) {
    const _Float16* ksrc = kbase + (size_t)nb * DH;
#if USE_ASYNC_LDS
#pragma unroll
    for (int s = 0; s < 2; ++s) {
      int seg = tid + s * 128;                       // 256 x 16 B
      async_cp16(ksrc + seg * 8, &Kl[buf][seg * 8]);
    }
#pragma unroll
    for (int s = 0; s < 2; ++s) {
      int seg = tid + s * 128;
      int row = seg >> 3, c16 = seg & 7;
      async_cp16(qbase + (size_t)row * MM + nb + c16 * 8,
                 &Ql[buf][row * 64 + c16 * 8]);
    }
#else
#pragma unroll
    for (int s = 0; s < 2; ++s) {
      int seg = tid + s * 128;
      *(uint4*)&Kl[buf][seg * 8] = *(const uint4*)(ksrc + seg * 8);
    }
#pragma unroll
    for (int s = 0; s < 2; ++s) {
      int seg = tid + s * 128;
      int row = seg >> 3, c16 = seg & 7;
      *(uint4*)&Ql[buf][row * 64 + c16 * 8] =
          *(const uint4*)(qbase + (size_t)row * MM + nb + c16 * 8);
    }
#endif
  };

  stage(0, 0);

  for (int it = 0; it < MM / 64; ++it) {
    const int cur = it & 1;
#if USE_ASYNC_LDS
    __builtin_amdgcn_s_wait_asynccnt(0);
#endif
    __syncthreads();                         // staged chunk visible block-wide
    if (it + 1 < MM / 64) stage(cur ^ 1, (it + 1) * 64);

    // ---- scores: 4 n-tiles of 16
    v8f s[4];
#pragma unroll
    for (int t = 0; t < 4; ++t) {
      Frag16 bk;
      const _Float16* kr = &Kl[cur][(t * 16 + (lane & 15)) * 32 + (lo ? 0 : 16)];
      bk.u[0] = *(const uint4*)(kr);
      bk.u[1] = *(const uint4*)(kr + 8);
      v8f z = {0,0,0,0,0,0,0,0};
      s[t] = wmma_f16(av.h, bk.h, z);
    }

    // ---- online softmax (base 2; scale folded into V)
    float corr[8];
#pragma unroll
    for (int r = 0; r < 8; ++r) {
      float t0 = fmaxf(fmaxf(s[0][r], s[1][r]), fmaxf(s[2][r], s[3][r]));
#pragma unroll
      for (int msk = 8; msk >= 1; msk >>= 1)
        t0 = fmaxf(t0, __shfl_xor(t0, msk, 32));
      float nm = fmaxf(rmax[r], t0);
      float cr = exp2f(rmax[r] - nm);
      rmax[r] = nm;
      Half4 p;
      float ps = 0.0f;
#pragma unroll
      for (int t = 0; t < 4; ++t) {
        float pv = exp2f(s[t][r] - nm);
        ps += pv;
        p.h[t] = (_Float16)pv;
      }
#pragma unroll
      for (int msk = 8; msk >= 1; msk >>= 1)
        ps += __shfl_xor(ps, msk, 32);
      rsum[r] = rsum[r] * cr + ps;
      corr[r] = cr;
      int mr = (lo ? 0 : 8) + r;
      // packed b64 store: positions col*4 + t  <->  n = t*16 + col
      *(uint2*)&P2[w][mr * 64 + (lane & 15) * 4] = p.u;
    }
#pragma unroll
    for (int r = 0; r < 8; ++r) { o0[r] *= corr[r]; o1[r] *= corr[r]; }

    asm volatile("s_wait_dscnt 0x0" ::: "memory");   // P stores -> P loads

    // ---- PV: two K=32 chunks (permuted n order matches Qd layout)
#pragma unroll
    for (int c = 0; c < 2; ++c) {
      Frag16 ap;
      const _Float16* pr = &P2[w][(lane & 15) * 64 + c * 32 + (lo ? 0 : 16)];
      ap.u[0] = *(const uint4*)(pr);
      ap.u[1] = *(const uint4*)(pr + 8);
      Frag16 bq0, bq1;
      const _Float16* q0 = &Ql[cur][(lane & 15) * 64 + c * 32 + (lo ? 0 : 16)];
      bq0.u[0] = *(const uint4*)(q0);
      bq0.u[1] = *(const uint4*)(q0 + 8);
      const _Float16* q1 = q0 + 16 * 64;
      bq1.u[0] = *(const uint4*)(q1);
      bq1.u[1] = *(const uint4*)(q1 + 8);
      o0 = wmma_f16(ap.h, bq0.h, o0);
      o1 = wmma_f16(ap.h, bq1.h, o1);
    }
    __syncthreads();                // all reads of buf[cur] done before restage
  }

  // ---- normalize and write out (B, C, M) fp32; rows m contiguous per lane
  float inv[8];
#pragma unroll
  for (int r = 0; r < 8; ++r) inv[r] = 1.0f / rsum[r];
  int m0 = mtile + (lo ? 0 : 8);
#pragma unroll
  for (int j = 0; j < 2; ++j) {
    int d = j * 16 + (lane & 15);
    int c = h * DH + d;
    const v8f& oa = j ? o1 : o0;
    F8 f;
#pragma unroll
    for (int r = 0; r < 8; ++r) f.f[r] = oa[r] * inv[r];
    float* dst = out + ((size_t)b * C_CH + c) * MM + m0;
    *(float4*)(dst)     = f.q[0];
    *(float4*)(dst + 4) = f.q[1];
  }
}

// ---------------------------------------------------------------------------
extern "C" void kernel_launch(void* const* d_in, const int* in_sizes, int n_in,
                              void* d_out, int out_size, void* d_ws, size_t ws_size,
                              hipStream_t stream) {
  const float* x  = (const float*)d_in[0];
  const float* Wv = (const float*)d_in[1];
  const float* Wz = (const float*)d_in[2];
  const float* Wq = (const float*)d_in[3];
  float* out = (float*)d_out;

  char* ws = (char*)d_ws;
  _Float16* Rt = (_Float16*)(ws);                            // 2 MB  (H,M,DH)
  _Float16* Vt = (_Float16*)(ws + (size_t) 2 * 1024 * 1024); // 4 MB  (B,H,M,DH)
  _Float16* Kt = (_Float16*)(ws + (size_t) 6 * 1024 * 1024); // 4 MB  (B,H,M,DH)
  _Float16* Qd = (_Float16*)(ws + (size_t)10 * 1024 * 1024); // 4 MB  (B,H,DH,M)

  pe_kernel<<<(HEADS * MM * DH) / 256, 256, 0, stream>>>(Rt);

  dim3 gp(MM / 16, 2, 1);                 // m-tiles x batch
  proj_kernel<<<gp, 256, 0, stream>>>(x, Wv, Wz, Wq, Rt, Vt, Kt, Qd);

  dim3 ga(MM / 64, HEADS, 2);             // m-blocks x heads x batch
  attn_kernel<<<ga, 128, 0, stream>>>(Vt, Kt, Qd, out);
}
